// ContextModule_56779467653251
// MI455X (gfx1250) — matrix operand order repaired and checked by reading, not compiled, not run on hardware
//
#include <hip/hip_runtime.h>
#include <math.h>

// ---------------------------------------------------------------------------
// ContextModule for MI455X (gfx1250): bf16 WMMA GEMMs + wave32 VALU glue.
// ---------------------------------------------------------------------------

typedef unsigned short u16;

typedef __attribute__((ext_vector_type(16))) __bf16 bfx16;
typedef __attribute__((ext_vector_type(8)))  __bf16 bfx8;
typedef __attribute__((ext_vector_type(8)))  float  fx8;

#define DMODEL 1024
#define DFFN   4096
#define NBATCH 32
#define NCTX   8192
#define KSEL   32
#define NHEAD  8
#define HDIM   128
#define SROWS  8224      // 32 q rows + 4096 sa rows + 4096 si rows (segment-major)
#define SA_OFF 32
#define SI_OFF 4128

__device__ __forceinline__ u16 f2bf(float f) {
    unsigned u = __float_as_uint(f);
    u += 0x7FFFu + ((u >> 16) & 1u);          // round-to-nearest-even
    return (u16)(u >> 16);
}
__device__ __forceinline__ float bf2f(u16 h) {
    return __uint_as_float(((unsigned)h) << 16);
}

union U4B8 { uint4 u; bfx8 b; };

// ---------------------------------------------------------------------------
// WMMA GEMM: out = act(A[M x Kd] * W + bias), W is Kd x N (row-major) or,
// if wT=1, stored N x Kd (used for qn @ ctx_n^T similarity GEMM).
// Block: 256 threads = 8 waves; block tile 128x128; K-step 32.
// Wave grid 4(M) x 2(N); each wave owns 32x64 = 2x4 WMMA 16x16 tiles.
// Requires: N % 128 == 0, Kd % 32 == 0 (true for all call sites).
// ---------------------------------------------------------------------------
#define LDS_STRIDE 48   // 32 + 16 pad; row pitch 96B (16B-aligned for b128 LDS ops)

__global__ __launch_bounds__(256)
void k_gemm_bf16(const u16* __restrict__ A, const u16* __restrict__ W,
                 const float* __restrict__ bias,
                 int M, int N, int Kd, int wT, int act,
                 float* __restrict__ outF, u16* __restrict__ outB)
{
    __shared__ u16 As[128][LDS_STRIDE];   // [m][k]
    __shared__ u16 Bs[128][LDS_STRIDE];   // [n][k]  (N-major so B-fragments are contiguous)

    const int tid  = threadIdx.x;
    const int lane = tid & 31;
    const int wave = tid >> 5;
    const int bm   = blockIdx.y * 128;
    const int bn   = blockIdx.x * 128;
    const int wmb  = (wave & 3) * 32;     // wave row offset inside block tile
    const int wnb  = (wave >> 2) * 64;    // wave col offset inside block tile

    fx8 acc[2][4];
#pragma unroll
    for (int i = 0; i < 2; ++i)
#pragma unroll
        for (int j = 0; j < 4; ++j)
#pragma unroll
            for (int e = 0; e < 8; ++e) acc[i][j][e] = 0.0f;

    for (int k0 = 0; k0 < Kd; k0 += 32) {
        // ---- stage A tile: 128 rows x 32 k (2 threads/row, 16 bf16 each) ----
        {
            const int r  = tid >> 1;
            const int c  = (tid & 1) * 16;
            const int gr = bm + r;
            uint4 v0 = make_uint4(0, 0, 0, 0), v1 = v0;
            if (gr < M) {
                const uint4* src = (const uint4*)(A + (size_t)gr * Kd + k0 + c);
                v0 = src[0]; v1 = src[1];
                if (k0 + 32 < Kd)  // prefetch next K tile (global_prefetch_b8)
                    __builtin_prefetch(A + (size_t)gr * Kd + k0 + 32 + c, 0, 1);
            }
            *(uint4*)&As[r][c]     = v0;
            *(uint4*)&As[r][c + 8] = v1;
        }
        // ---- stage W tile into N-major LDS ----
        if (wT) {  // W stored N x Kd: straight row copy
            const int n = tid >> 1;
            const int c = (tid & 1) * 16;
            const uint4* src = (const uint4*)(W + (size_t)(bn + n) * Kd + k0 + c);
            *(uint4*)&Bs[n][c]     = src[0];
            *(uint4*)&Bs[n][c + 8] = src[1];
            if (k0 + 32 < Kd)
                __builtin_prefetch(W + (size_t)(bn + n) * Kd + k0 + 32 + c, 0, 1);
        } else {   // W stored Kd x N: transpose on the way into LDS
            const int k  = tid >> 3;          // 0..31
            const int n0 = (tid & 7) * 16;    // 0..112
            const u16* src = W + (size_t)(k0 + k) * N + bn + n0;
#pragma unroll
            for (int j = 0; j < 16; ++j) Bs[n0 + j][k] = src[j];
            if (k0 + 32 < Kd)
                __builtin_prefetch(W + (size_t)(k0 + 32 + k) * N + bn + n0, 0, 1);
        }
        __syncthreads();

        // ---- fragments per ISA 16-bit layouts ----
        bfx16 afr[2], bfr[4];
        {
            const int koff = (lane >> 4) * 8;     // A: lanes16-31 hold K 8..15 / 24..31
#pragma unroll
            for (int i = 0; i < 2; ++i) {
                const int r = wmb + i * 16 + (lane & 15);
                U4B8 lo, hi;
                lo.u = *(const uint4*)&As[r][koff];
                hi.u = *(const uint4*)&As[r][koff + 16];
                afr[i] = __builtin_shufflevector(lo.b, hi.b,
                         0,1,2,3,4,5,6,7,8,9,10,11,12,13,14,15);
            }
            const int kb = (lane >> 4) * 8;       // B: lanes16-31 hold K 8..15 / 24..31
#pragma unroll
            for (int j = 0; j < 4; ++j) {
                const int n = wnb + j * 16 + (lane & 15);
                U4B8 lo, hi;
                lo.u = *(const uint4*)&Bs[n][kb];
                hi.u = *(const uint4*)&Bs[n][kb + 16];
                bfr[j] = __builtin_shufflevector(lo.b, hi.b,
                         0,1,2,3,4,5,6,7,8,9,10,11,12,13,14,15);
            }
        }
#pragma unroll
        for (int i = 0; i < 2; ++i)
#pragma unroll
            for (int j = 0; j < 4; ++j)
                acc[i][j] = __builtin_amdgcn_wmma_f32_16x16x32_bf16(
                    false, afr[i], false, bfr[j], (short)0, acc[i][j], false, false);
        __syncthreads();
    }

    // ---- epilogue: C/D layout — VGPR e: row = (lane>=16)*8 + e, col = lane&15 ----
    const int rofs = (lane >> 4) * 8;
#pragma unroll
    for (int i = 0; i < 2; ++i) {
#pragma unroll
        for (int j = 0; j < 4; ++j) {
            const int col = bn + wnb + j * 16 + (lane & 15);
            const float bval = bias ? bias[col] : 0.0f;
#pragma unroll
            for (int e = 0; e < 8; ++e) {
                const int row = bm + wmb + i * 16 + rofs + e;
                if (row < M) {
                    float v = acc[i][j][e] + bval;
                    if (act == 1) {  // tanh-gelu (JAX default approximate=True)
                        v = 0.5f * v * (1.0f + tanhf(0.7978845608028654f *
                                                     (v + 0.044715f * v * v * v)));
                    }
                    const size_t o = (size_t)row * N + col;
                    if (outF) outF[o] = v;
                    if (outB) outB[o] = f2bf(v);
                }
            }
        }
    }
}

// ---------------------------------------------------------------------------
// Row LayerNorm (D=1024), one block per row.
// ---------------------------------------------------------------------------
__global__ __launch_bounds__(256)
void k_layernorm(const float* __restrict__ x, const float* __restrict__ g,
                 const float* __restrict__ b,
                 float* __restrict__ outF, u16* __restrict__ outB)
{
    __shared__ float red[256];
    const int tid = threadIdx.x;
    const float* xr = x + (size_t)blockIdx.x * DMODEL;
    float v[4]; float s = 0.0f;
#pragma unroll
    for (int i = 0; i < 4; ++i) { v[i] = xr[tid + 256 * i]; s += v[i]; }
    red[tid] = s; __syncthreads();
    for (int st = 128; st > 0; st >>= 1) {
        if (tid < st) red[tid] += red[tid + st];
        __syncthreads();
    }
    const float mean = red[0] * (1.0f / DMODEL);
    __syncthreads();
    float ss = 0.0f;
#pragma unroll
    for (int i = 0; i < 4; ++i) { const float d = v[i] - mean; ss += d * d; }
    red[tid] = ss; __syncthreads();
    for (int st = 128; st > 0; st >>= 1) {
        if (tid < st) red[tid] += red[tid + st];
        __syncthreads();
    }
    const float rstd = rsqrtf(red[0] * (1.0f / DMODEL) + 1e-5f);
#pragma unroll
    for (int i = 0; i < 4; ++i) {
        const int d = tid + 256 * i;
        const float o = (v[i] - mean) * rstd * g[d] + b[d];
        const size_t idx = (size_t)blockIdx.x * DMODEL + d;
        if (outF) outF[idx] = o;
        if (outB) outB[idx] = f2bf(o);
    }
}

// ---------------------------------------------------------------------------
// Row L2-normalize (D=1024): x / max(||x||, eps)
// ---------------------------------------------------------------------------
__global__ __launch_bounds__(256)
void k_l2norm(const float* __restrict__ x, float eps,
              float* __restrict__ outF, u16* __restrict__ outB)
{
    __shared__ float red[256];
    const int tid = threadIdx.x;
    const float* xr = x + (size_t)blockIdx.x * DMODEL;
    float v[4]; float ss = 0.0f;
#pragma unroll
    for (int i = 0; i < 4; ++i) { v[i] = xr[tid + 256 * i]; ss += v[i] * v[i]; }
    red[tid] = ss; __syncthreads();
    for (int st = 128; st > 0; st >>= 1) {
        if (tid < st) red[tid] += red[tid + st];
        __syncthreads();
    }
    const float scale = 1.0f / fmaxf(sqrtf(red[0]), eps);
#pragma unroll
    for (int i = 0; i < 4; ++i) {
        const int d = tid + 256 * i;
        const float o = v[i] * scale;
        const size_t idx = (size_t)blockIdx.x * DMODEL + d;
        if (outF) outF[idx] = o;
        if (outB) outB[idx] = f2bf(o);
    }
}

__global__ void k_f32_to_bf16(const float* __restrict__ in, u16* __restrict__ out, long n)
{
    long i = (long)blockIdx.x * blockDim.x + threadIdx.x;
    const long stride = (long)gridDim.x * blockDim.x;
    for (; i < n; i += stride) out[i] = f2bf(in[i]);
}

// ---------------------------------------------------------------------------
// Top-32 indices per batch row of sim[32][8192] (set-selection; attention
// output is permutation-invariant over the selected keys).
// ---------------------------------------------------------------------------
__global__ __launch_bounds__(256)
void k_topk(const float* __restrict__ sim, int* __restrict__ idx)
{
    __shared__ float vals[NCTX];
    __shared__ float rv[256];
    __shared__ int   ri[256];
    const int b = blockIdx.x, tid = threadIdx.x;
    for (int i = tid; i < NCTX; i += 256) vals[i] = sim[(size_t)b * NCTX + i];
    __syncthreads();
    for (int t = 0; t < KSEL; ++t) {
        float bestv = -1e30f; int besti = 0;
        for (int i = tid; i < NCTX; i += 256) {
            const float v = vals[i];
            if (v > bestv) { bestv = v; besti = i; }
        }
        rv[tid] = bestv; ri[tid] = besti; __syncthreads();
        for (int st = 128; st > 0; st >>= 1) {
            if (tid < st && rv[tid + st] > rv[tid]) { rv[tid] = rv[tid + st]; ri[tid] = ri[tid + st]; }
            __syncthreads();
        }
        if (tid == 0) { idx[b * KSEL + t] = ri[0]; vals[ri[0]] = -1e30f; }
        __syncthreads();
    }
}

// Gather selected context rows (post-LN, f32) into bf16 (B*K) x D
__global__ __launch_bounds__(256)
void k_gather(const float* __restrict__ ctx, const int* __restrict__ idx,
              u16* __restrict__ out)
{
    const int r = blockIdx.x;           // 0..1023 == b*32 + k
    const int src = idx[r];
    for (int i = threadIdx.x; i < DMODEL; i += 256)
        out[(size_t)r * DMODEL + i] = f2bf(ctx[(size_t)src * DMODEL + i]);
}

// ---------------------------------------------------------------------------
// 32-key attention: one wave32 per (b,h,s); lane == key index.
// ---------------------------------------------------------------------------
__global__ __launch_bounds__(256)
void k_attn(const u16* __restrict__ qh, const u16* __restrict__ kh,
            const u16* __restrict__ vh, const float* __restrict__ beta,
            u16* __restrict__ out)
{
    const int w = blockIdx.x * 8 + (threadIdx.x >> 5);
    const int lane = threadIdx.x & 31;
    const int b = w / (NHEAD * 257);
    const int rem = w - b * (NHEAD * 257);
    const int h = rem / 257;
    const int s = rem - h * 257;
    const int row = (s == 0) ? b
                  : (s <= 128 ? SA_OFF + b * 128 + (s - 1)
                              : SI_OFF + b * 128 + (s - 129));
    const u16* qp = qh + (size_t)row * DMODEL + h * HDIM;
    const u16* kp = kh + (size_t)(b * KSEL + lane) * DMODEL + h * HDIM;
    float dot = 0.0f;
#pragma unroll 4
    for (int d = 0; d < HDIM; ++d) dot += bf2f(qp[d]) * bf2f(kp[d]);
    const float sc = dot * beta[0] * 0.08838834764831845f;  // 1/sqrt(128)
    float m = sc;
#pragma unroll
    for (int o = 16; o > 0; o >>= 1) m = fmaxf(m, __shfl_xor(m, o));
    const float e = __expf(sc - m);
    float sum = e;
#pragma unroll
    for (int o = 16; o > 0; o >>= 1) sum += __shfl_xor(sum, o);
    const float attn = e / sum;
    float o0 = 0.f, o1 = 0.f, o2 = 0.f, o3 = 0.f;
    for (int k2 = 0; k2 < KSEL; ++k2) {
        const float a = __shfl(attn, k2);
        const u16* vp = vh + (size_t)(b * KSEL + k2) * DMODEL + h * HDIM;
        o0 += a * bf2f(vp[lane]);
        o1 += a * bf2f(vp[lane + 32]);
        o2 += a * bf2f(vp[lane + 64]);
        o3 += a * bf2f(vp[lane + 96]);
    }
    u16* op = out + (size_t)row * DMODEL + h * HDIM;
    op[lane]      = f2bf(o0);
    op[lane + 32] = f2bf(o1);
    op[lane + 64] = f2bf(o2);
    op[lane + 96] = f2bf(o3);
}

// x = x + sigmoid(gate[d]) * (hop - sp), segment-major row mapping
__global__ __launch_bounds__(256)
void k_residual(float* __restrict__ q, float* __restrict__ sa, float* __restrict__ si,
                const float* __restrict__ hop, const float* __restrict__ sp,
                const float* __restrict__ qgate, const float* __restrict__ sgate)
{
    const size_t i = (size_t)blockIdx.x * blockDim.x + threadIdx.x;
    if (i >= (size_t)SROWS * DMODEL) return;
    const int r = (int)(i >> 10), d = (int)(i & 1023);
    const float delta = hop[i] - sp[i];
    if (r < SA_OFF) {
        const float g = 1.0f / (1.0f + __expf(-qgate[d]));
        q[(size_t)r * DMODEL + d] += g * delta;
    } else {
        const float g = 1.0f / (1.0f + __expf(-sgate[d]));
        if (r < SI_OFF) sa[(size_t)(r - SA_OFF) * DMODEL + d] += g * delta;
        else            si[(size_t)(r - SI_OFF) * DMODEL + d] += g * delta;
    }
}

// x = x + f (FFN residual), same mapping
__global__ __launch_bounds__(256)
void k_addseg(float* __restrict__ q, float* __restrict__ sa, float* __restrict__ si,
              const float* __restrict__ f)
{
    const size_t i = (size_t)blockIdx.x * blockDim.x + threadIdx.x;
    if (i >= (size_t)SROWS * DMODEL) return;
    const int r = (int)(i >> 10), d = (int)(i & 1023);
    if (r < SA_OFF)      q[(size_t)r * DMODEL + d] += f[i];
    else if (r < SI_OFF) sa[(size_t)(r - SA_OFF) * DMODEL + d] += f[i];
    else                 si[(size_t)(r - SI_OFF) * DMODEL + d] += f[i];
}

// ---------------------------------------------------------------------------
// Host orchestration
// ---------------------------------------------------------------------------
static inline void gemm(hipStream_t st, const u16* A, const u16* W, const float* bias,
                        int M, int N, int Kd, int wT, int act, float* oF, u16* oB)
{
    dim3 grid(N / 128, (M + 127) / 128);
    k_gemm_bf16<<<grid, 256, 0, st>>>(A, W, bias, M, N, Kd, wT, act, oF, oB);
}

extern "C" void kernel_launch(void* const* d_in, const int* in_sizes, int n_in,
                              void* d_out, int out_size, void* d_ws, size_t ws_size,
                              hipStream_t stream)
{
    const float* query  = (const float*)d_in[0];
    const float* sup_a  = (const float*)d_in[1];
    const float* sup_i  = (const float*)d_in[2];
    const float* context= (const float*)d_in[3];
    const float* pre_g  = (const float*)d_in[4];
    const float* pre_b  = (const float*)d_in[5];
    const float* Wqp = (const float*)d_in[6];  const float* bqp = (const float*)d_in[7];
    const float* Wap = (const float*)d_in[8];  const float* bap = (const float*)d_in[9];
    const float* Wip = (const float*)d_in[10]; const float* bip = (const float*)d_in[11];
    const float* Wq  = (const float*)d_in[12]; const float* bq  = (const float*)d_in[13];
    const float* Wk  = (const float*)d_in[14]; const float* bk  = (const float*)d_in[15];
    const float* Wv  = (const float*)d_in[16]; const float* bv  = (const float*)d_in[17];
    const float* Wo  = (const float*)d_in[18]; const float* bo  = (const float*)d_in[19];
    const float* hop_beta = (const float*)d_in[20];
    const float* qgate = (const float*)d_in[21];
    const float* sgate = (const float*)d_in[22];
    const float* ffn_g = (const float*)d_in[23];
    const float* ffn_b = (const float*)d_in[24];
    const float* W1 = (const float*)d_in[25]; const float* b1 = (const float*)d_in[26];
    const float* W2 = (const float*)d_in[27]; const float* b2 = (const float*)d_in[28];

    (void)in_sizes; (void)n_in; (void)out_size; (void)ws_size;

    // ---- workspace arena (256B aligned) ----
    char* base = (char*)d_ws; size_t off = 0;
    auto alloc = [&](size_t bytes) -> void* {
        void* p = base + off;
        off = (off + bytes + 255) & ~(size_t)255;
        return p;
    };
    const size_t DD = (size_t)DMODEL;
    // persistent state
    float* q   = (float*)alloc(32 * DD * 4);
    float* sa  = (float*)alloc((size_t)4096 * DD * 4);
    float* si  = (float*)alloc((size_t)4096 * DD * 4);
    float* ctxn    = (float*)alloc((size_t)NCTX * DD * 4);   // LayerNormed context
    u16* ctxn_bf   = (u16*)alloc((size_t)NCTX * DD * 2);     // l2-normalized, bf16
    u16* x_bf      = (u16*)alloc((size_t)SROWS * DD * 2);    // bf16 mirror of q|sa|si
    // bf16 weights
    u16* wqp_bf = (u16*)alloc(DD * DD * 2);
    u16* wap_bf = (u16*)alloc(DD * DD * 2);
    u16* wip_bf = (u16*)alloc(DD * DD * 2);
    u16* wq_bf  = (u16*)alloc(DD * DD * 2);
    u16* wk_bf  = (u16*)alloc(DD * DD * 2);
    u16* wv_bf  = (u16*)alloc(DD * DD * 2);
    u16* wo_bf  = (u16*)alloc(DD * DD * 2);
    u16* w1_bf  = (u16*)alloc(DD * (size_t)DFFN * 2);
    u16* w2_bf  = (u16*)alloc((size_t)DFFN * DD * 2);
    // per-step scratch
    u16*   qn_bf   = (u16*)alloc(32 * DD * 2);
    float* sim     = (float*)alloc((size_t)NBATCH * NCTX * 4);
    int*   idx     = (int*)alloc(NBATCH * KSEL * 4);
    u16*   ctxk_bf = (u16*)alloc((size_t)NBATCH * KSEL * DD * 2);
    float* s_p     = (float*)alloc((size_t)SROWS * DD * 4);
    u16*   s_p_bf  = (u16*)alloc((size_t)SROWS * DD * 2);
    u16*   qh_bf   = (u16*)alloc((size_t)SROWS * DD * 2);
    u16*   kh_bf   = (u16*)alloc((size_t)NBATCH * KSEL * DD * 2);
    u16*   vh_bf   = (u16*)alloc((size_t)NBATCH * KSEL * DD * 2);
    u16*   attn_bf = (u16*)alloc((size_t)SROWS * DD * 2);
    float* hop     = (float*)alloc((size_t)SROWS * DD * 4);  // reused for FFN output
    u16*   xn_bf   = (u16*)alloc((size_t)SROWS * DD * 2);
    u16*   h1_bf   = (u16*)alloc((size_t)SROWS * (size_t)DFFN * 2);

    // ---- one-time preprocessing ----
    k_layernorm<<<32,   256, 0, stream>>>(query,   pre_g, pre_b, q,    nullptr);
    k_layernorm<<<4096, 256, 0, stream>>>(sup_a,   pre_g, pre_b, sa,   nullptr);
    k_layernorm<<<4096, 256, 0, stream>>>(sup_i,   pre_g, pre_b, si,   nullptr);
    k_layernorm<<<NCTX, 256, 0, stream>>>(context, pre_g, pre_b, ctxn, nullptr);
    k_l2norm<<<NCTX, 256, 0, stream>>>(ctxn, 1e-12f, nullptr, ctxn_bf);

    k_f32_to_bf16<<<4096,  256, 0, stream>>>(Wqp, wqp_bf, (long)DD * DD);
    k_f32_to_bf16<<<4096,  256, 0, stream>>>(Wap, wap_bf, (long)DD * DD);
    k_f32_to_bf16<<<4096,  256, 0, stream>>>(Wip, wip_bf, (long)DD * DD);
    k_f32_to_bf16<<<4096,  256, 0, stream>>>(Wq,  wq_bf,  (long)DD * DD);
    k_f32_to_bf16<<<4096,  256, 0, stream>>>(Wk,  wk_bf,  (long)DD * DD);
    k_f32_to_bf16<<<4096,  256, 0, stream>>>(Wv,  wv_bf,  (long)DD * DD);
    k_f32_to_bf16<<<4096,  256, 0, stream>>>(Wo,  wo_bf,  (long)DD * DD);
    k_f32_to_bf16<<<16384, 256, 0, stream>>>(W1,  w1_bf,  (long)DD * DFFN);
    k_f32_to_bf16<<<16384, 256, 0, stream>>>(W2,  w2_bf,  (long)DFFN * DD);

    const int elemBlocks = (int)(((size_t)SROWS * DMODEL) / 256);  // exact

    // ---- 3 refinement steps ----
    for (int step = 0; step < 3; ++step) {
        // bf16 mirrors of current state (segment-major S layout)
        k_f32_to_bf16<<<128,   256, 0, stream>>>(q,  x_bf,                 (long)32 * DD);
        k_f32_to_bf16<<<16384, 256, 0, stream>>>(sa, x_bf + SA_OFF * DD,   (long)4096 * DD);
        k_f32_to_bf16<<<16384, 256, 0, stream>>>(si, x_bf + SI_OFF * DD,   (long)4096 * DD);

        // retrieval: sim = l2n(q) @ ctx_n^T ; top-32 ; gather
        k_l2norm<<<NBATCH, 256, 0, stream>>>(q, 1e-12f, nullptr, qn_bf);
        gemm(stream, qn_bf, ctxn_bf, nullptr, NBATCH, NCTX, DMODEL, /*wT=*/1, 0, sim, nullptr);
        k_topk<<<NBATCH, 256, 0, stream>>>(sim, idx);
        k_gather<<<NBATCH * KSEL, 256, 0, stream>>>(ctxn, idx, ctxk_bf);

        // projections into s = [q_p | sa_p | si_p]
        gemm(stream, x_bf,               wqp_bf, bqp, 32,   DMODEL, DMODEL, 0, 0,
             s_p,               s_p_bf);
        gemm(stream, x_bf + SA_OFF * DD, wap_bf, bap, 4096, DMODEL, DMODEL, 0, 0,
             s_p + SA_OFF * DD, s_p_bf + SA_OFF * DD);
        gemm(stream, x_bf + SI_OFF * DD, wip_bf, bip, 4096, DMODEL, DMODEL, 0, 0,
             s_p + SI_OFF * DD, s_p_bf + SI_OFF * DD);

        // hopfield: Q/K/V projections, 32-key attention, output projection
        gemm(stream, s_p_bf,  wq_bf, bq, SROWS,          DMODEL, DMODEL, 0, 0, nullptr, qh_bf);
        gemm(stream, ctxk_bf, wk_bf, bk, NBATCH * KSEL,  DMODEL, DMODEL, 0, 0, nullptr, kh_bf);
        gemm(stream, ctxk_bf, wv_bf, bv, NBATCH * KSEL,  DMODEL, DMODEL, 0, 0, nullptr, vh_bf);
        k_attn<<<(NBATCH * NHEAD * 257) / 8, 256, 0, stream>>>(qh_bf, kh_bf, vh_bf,
                                                               hop_beta, attn_bf);
        gemm(stream, attn_bf, wo_bf, bo, SROWS, DMODEL, DMODEL, 0, 0, hop, nullptr);

        // gated residual
        k_residual<<<elemBlocks, 256, 0, stream>>>(q, sa, si, hop, s_p, qgate, sgate);

        // FFN block: LN -> W1+gelu -> W2 -> residual add
        k_layernorm<<<32,   256, 0, stream>>>(q,  ffn_g, ffn_b, nullptr, xn_bf);
        k_layernorm<<<4096, 256, 0, stream>>>(sa, ffn_g, ffn_b, nullptr, xn_bf + SA_OFF * DD);
        k_layernorm<<<4096, 256, 0, stream>>>(si, ffn_g, ffn_b, nullptr, xn_bf + SI_OFF * DD);
        gemm(stream, xn_bf, w1_bf, b1, SROWS, DFFN,   DMODEL, 0, /*gelu*/1, nullptr, h1_bf);
        gemm(stream, h1_bf, w2_bf, b2, SROWS, DMODEL, DFFN,   0, 0, hop, nullptr);
        k_addseg<<<elemBlocks, 256, 0, stream>>>(q, sa, si, hop);
    }

    // ---- final L2 normalization into d_out (q | sa | si, f32) ----
    float* out = (float*)d_out;
    k_l2norm<<<32,   256, 0, stream>>>(q,  1e-8f, out,                          nullptr);
    k_l2norm<<<4096, 256, 0, stream>>>(sa, 1e-8f, out + 32 * DD,                nullptr);
    k_l2norm<<<4096, 256, 0, stream>>>(si, 1e-8f, out + (32 + 4096) * DD,       nullptr);
}
